// MultiHeadAttentionWithCLSToken_944892805457
// MI455X (gfx1250) — matrix operand order repaired and checked by reading, not compiled, not run on hardware
//
#include <hip/hip_runtime.h>
#include <hip/hip_bf16.h>

#define BATCH  32
#define LTOT   2049
#define EMB    1024
#define NH     16
#define HD     64
#define NSPLIT 16
#define SCHUNK 128    // 2048 / NSPLIT
#define XTSTR  (SCHUNK + 8)   // padded LDS stride for transposed x image

typedef __attribute__((ext_vector_type(16))) __bf16 v16bf;
typedef __attribute__((ext_vector_type(8)))  float  v8f;

// dynamic LDS layout (bytes):
//   scores : NH*SCHUNK floats                 = 8192
//   red    : NH*16 floats                     = 1024
//   pbf    : NH*SCHUNK bf16                   = 4096
//   xT     : EMB*XTSTR bf16 (transposed x)    = 278528
#define SMEM_BYTES ((NH*SCHUNK + NH*16) * 4 + (NH*SCHUNK) * 2 + (EMB*XTSTR) * 2)

// ---------------------------------------------------------------------------
// K1: q[b] = Wq @ cls[b] + bq ;  r[b,h,:] = sum_d q[b,h,d] * Wk[h*64+d, :]
// ---------------------------------------------------------------------------
__global__ __launch_bounds__(256) void qr_kernel(
    const float* __restrict__ x, const float* __restrict__ Wq,
    const float* __restrict__ bq, const float* __restrict__ Wk,
    float* __restrict__ r) {
  const int b = blockIdx.x, t = threadIdx.x;
  __shared__ float cls[EMB];
  __shared__ float q[EMB];
  for (int e = t; e < EMB; e += 256) cls[e] = x[(size_t)b * LTOT * EMB + e];
  __syncthreads();
  for (int k = 0; k < 4; ++k) {
    int i = t + k * 256;
    float acc = bq[i];
    const float* wr = Wq + (size_t)i * EMB;
    for (int e = 0; e < EMB; ++e) acc += wr[e] * cls[e];
    q[i] = acc;
  }
  __syncthreads();
  for (int h = 0; h < NH; ++h) {
    const float* qh = q + h * HD;
    for (int k = 0; k < 4; ++k) {
      int e = t + k * 256;
      float acc = 0.f;
      const float* wk = Wk + (size_t)h * HD * EMB + e;
      for (int d = 0; d < HD; ++d) acc += qh[d] * wk[(size_t)d * EMB];
      r[(size_t)(b * NH + h) * EMB + e] = acc;
    }
  }
}

// ---------------------------------------------------------------------------
// K2: per (b, split of 128 rows): WMMA scores -> chunk softmax partial ->
//     WMMA ctx partial.  x is read from HBM exactly once: phase 1 converts it
//     to bf16 and deposits a transposed image xT[e][s] in LDS, which phase 2
//     consumes with contiguous 32B reads as the WMMA B operand.
// ---------------------------------------------------------------------------
__global__ __launch_bounds__(256) void flash_kernel(
    const float* __restrict__ x, const float* __restrict__ r,
    float* __restrict__ parts, float* __restrict__ ml) {
  const int split = blockIdx.x, b = blockIdx.y;
  const int tid = threadIdx.x, lane = tid & 31, wave = tid >> 5;
  const int mrow = lane & 15;
  const bool lo = lane < 16;

  extern __shared__ __align__(16) char smem[];
  float*  scores = (float*)smem;                       // [NH][SCHUNK]
  float*  red    = scores + NH * SCHUNK;               // [NH][16]
  __bf16* pbf    = (__bf16*)(red + NH * 16);           // [NH][SCHUNK]
  __bf16* xT     = pbf + NH * SCHUNK;                  // [EMB][XTSTR]

  const int sbase = split * SCHUNK;
  const int aoff = lo ? 0 : 8;     // (unused) A now uses contiguous K via ko, matching B
  const int ko   = lo ? 0 : 16;    // contiguous K order used for A and B: K=0..15 (lanes 0-15), 16..31 (16-31)
  const int hrow = lo ? 0 : 8;     // C layout: VGPR j -> M=j (lanes 0-15) / M=j+8 (16-31)

  // ---- phase 1: scores[h][s] = (r[b,h,:] . x[b, 1+sbase+s, :]) / 8 --------
  {
    const float* rrow = r + (size_t)(b * NH + mrow) * EMB;
    const int s0 = wave * 16;      // this wave owns one 16-column s-tile
    const int sl = s0 + mrow;      // the x row this lane streams
    const float* x0 = x + ((size_t)b * LTOT + 1 + sbase + sl) * EMB + ko;
    v8f c0 = {};
    for (int e = 0; e < EMB; e += 32) {
      __builtin_prefetch(x0 + e + 256, 0, 0);   // global_prefetch_b8
      v16bf a, b0;
      #pragma unroll
      for (int i = 0; i < 16; ++i) a[i] = (__bf16)rrow[e + ko + i];
      #pragma unroll
      for (int i = 0; i < 16; ++i) b0[i] = (__bf16)x0[e + i];
      // deposit transposed bf16 image for phase 2
      #pragma unroll
      for (int i = 0; i < 16; ++i) xT[(e + ko + i) * XTSTR + sl] = b0[i];
      c0 = __builtin_amdgcn_wmma_f32_16x16x32_bf16(false, a, false, b0, (short)0, c0, false, false);
    }
    #pragma unroll
    for (int j = 0; j < 8; ++j)
      scores[(hrow + j) * SCHUNK + s0 + mrow] = c0[j] * 0.125f;
  }
  __syncthreads();

  // ---- chunk-local softmax partial per head -------------------------------
  const int h = tid >> 4, j = tid & 15;
  {
    float lm = -3.4e38f;
    for (int s = j; s < SCHUNK; s += 16) lm = fmaxf(lm, scores[h * SCHUNK + s]);
    red[h * 16 + j] = lm;
    __syncthreads();
    float m = red[h * 16];
    #pragma unroll
    for (int k = 1; k < 16; ++k) m = fmaxf(m, red[h * 16 + k]);
    __syncthreads();
    float ls = 0.f;
    for (int s = j; s < SCHUNK; s += 16) {
      float p = __expf(scores[h * SCHUNK + s] - m);
      ls += p;
      pbf[h * SCHUNK + s] = (__bf16)p;
    }
    red[h * 16 + j] = ls;
    __syncthreads();
    float l = 0.f;
    #pragma unroll
    for (int k = 0; k < 16; ++k) l += red[h * 16 + k];
    if (j == 0) {
      size_t idx = ((size_t)(b * NSPLIT + split) * NH + h) * 2;
      ml[idx] = m; ml[idx + 1] = l;
    }
  }
  __syncthreads();

  // ---- phase 2: ctx_part[h][e] = sum_s pbf[h][s] * xT[e][s] ---------------
  {
    for (int tt = 0; tt < 8; ++tt) {
      const int e0 = (wave * 8 + tt) * 16;
      const __bf16* xrow = xT + (size_t)(e0 + mrow) * XTSTR + ko;  // contiguous in s
      v8f c = {};
      for (int sk = 0; sk < SCHUNK; sk += 32) {
        v16bf a, bb;
        #pragma unroll
        for (int i = 0; i < 16; ++i) a[i] = pbf[mrow * SCHUNK + sk + ko + i];
        #pragma unroll
        for (int i = 0; i < 16; ++i) bb[i] = xrow[sk + i];
        c = __builtin_amdgcn_wmma_f32_16x16x32_bf16(false, a, false, bb, (short)0, c, false, false);
      }
      float* prt = parts + ((size_t)(b * NSPLIT + split) * NH) * EMB + e0 + mrow;
      #pragma unroll
      for (int jj = 0; jj < 8; ++jj)
        prt[(size_t)(hrow + jj) * EMB] = c[jj];
    }
  }
}

// ---------------------------------------------------------------------------
// K3: merge split partials -> ctx[b,h,:]
// ---------------------------------------------------------------------------
__global__ __launch_bounds__(256) void combine_kernel(
    const float* __restrict__ parts, const float* __restrict__ ml,
    float* __restrict__ ctx) {
  const int b = blockIdx.x, t = threadIdx.x;
  __shared__ float coef[NSPLIT][NH];
  __shared__ float mA[NSPLIT][NH], lA[NSPLIT][NH];
  {
    int sp = t >> 4, hh = t & 15;      // 256 threads == NSPLIT*NH slots
    size_t idx = ((size_t)(b * NSPLIT + sp) * NH + hh) * 2;
    mA[sp][hh] = ml[idx]; lA[sp][hh] = ml[idx + 1];
  }
  __syncthreads();
  if (t < NH) {
    float M = mA[0][t];
    for (int sp = 1; sp < NSPLIT; ++sp) M = fmaxf(M, mA[sp][t]);
    float L = 0.f;
    for (int sp = 0; sp < NSPLIT; ++sp) L += __expf(mA[sp][t] - M) * lA[sp][t];
    float inv = 1.f / L;
    for (int sp = 0; sp < NSPLIT; ++sp) coef[sp][t] = __expf(mA[sp][t] - M) * inv;
  }
  __syncthreads();
  for (int h = 0; h < NH; ++h)
    for (int e = t; e < EMB; e += 256) {
      float acc = 0.f;
      for (int sp = 0; sp < NSPLIT; ++sp)
        acc += coef[sp][h] * parts[((size_t)(b * NSPLIT + sp) * NH + h) * EMB + e];
      ctx[((size_t)b * NH + h) * EMB + e] = acc;
    }
}

// ---------------------------------------------------------------------------
// K4: attn = Wv.ctx + bv ; out = Wo.attn + bo + cls ; LayerNorm
// ---------------------------------------------------------------------------
__global__ __launch_bounds__(256) void final_kernel(
    const float* __restrict__ x, const float* __restrict__ ctx,
    const float* __restrict__ Wv, const float* __restrict__ bv,
    const float* __restrict__ Wo, const float* __restrict__ bo,
    const float* __restrict__ g, const float* __restrict__ bb,
    float* __restrict__ out) {
  const int b = blockIdx.x, t = threadIdx.x;
  __shared__ float attn[EMB];
  __shared__ float y[EMB];
  __shared__ float red[256];
  for (int k = 0; k < 4; ++k) {
    int i = t + k * 256; int h = i >> 6;
    float acc = bv[i];
    const float* wr = Wv + (size_t)i * EMB;
    const float* cr = ctx + ((size_t)b * NH + h) * EMB;
    for (int e = 0; e < EMB; ++e) acc += wr[e] * cr[e];
    attn[i] = acc;
  }
  __syncthreads();
  for (int k = 0; k < 4; ++k) {
    int i = t + k * 256;
    float acc = bo[i] + x[(size_t)b * LTOT * EMB + i];   // + cls residual
    const float* wr = Wo + (size_t)i * EMB;
    for (int e = 0; e < EMB; ++e) acc += wr[e] * attn[e];
    y[i] = acc;
  }
  __syncthreads();
  float s = 0.f;
  for (int k = 0; k < 4; ++k) s += y[t + k * 256];
  red[t] = s; __syncthreads();
  for (int o = 128; o; o >>= 1) { if (t < o) red[t] += red[t + o]; __syncthreads(); }
  float mean = red[0] * (1.0f / EMB);
  __syncthreads();
  s = 0.f;
  for (int k = 0; k < 4; ++k) { float d = y[t + k * 256] - mean; s += d * d; }
  red[t] = s; __syncthreads();
  for (int o = 128; o; o >>= 1) { if (t < o) red[t] += red[t + o]; __syncthreads(); }
  float rstd = rsqrtf(red[0] * (1.0f / EMB) + 1e-5f);
  for (int k = 0; k < 4; ++k) {
    int i = t + k * 256;
    out[(size_t)b * EMB + i] = (y[i] - mean) * rstd * g[i] + bb[i];
  }
}

// ---------------------------------------------------------------------------
extern "C" void kernel_launch(void* const* d_in, const int* in_sizes, int n_in,
                              void* d_out, int out_size, void* d_ws, size_t ws_size,
                              hipStream_t stream) {
  const float* x  = (const float*)d_in[0];
  const float* Wq = (const float*)d_in[1];
  const float* bq = (const float*)d_in[2];
  const float* Wk = (const float*)d_in[3];
  // d_in[4] = bk: constant over s -> softmax-invariant, intentionally unused
  const float* Wv = (const float*)d_in[5];
  const float* bv = (const float*)d_in[6];
  const float* Wo = (const float*)d_in[7];
  const float* bo = (const float*)d_in[8];
  const float* g  = (const float*)d_in[9];
  const float* be = (const float*)d_in[10];
  float* out = (float*)d_out;
  float* ws = (float*)d_ws;
  (void)in_sizes; (void)n_in; (void)out_size; (void)ws_size;

  float* r     = ws;                      // 32*16*1024             = 524288 f
  float* parts = r + 524288;              // 32*16*16*1024          = 8388608 f
  float* ml    = parts + 8388608;         // 32*16*16*2             = 16384 f
  float* ctx   = ml + 16384;              // 32*16*1024             = 524288 f
                                          // total ~36 MB of ws

  qr_kernel<<<BATCH, 256, 0, stream>>>(x, Wq, bq, Wk, r);
  flash_kernel<<<dim3(NSPLIT, BATCH), 256, SMEM_BYTES, stream>>>(x, r, parts, ml);
  combine_kernel<<<BATCH, 256, 0, stream>>>(parts, ml, ctx);
  final_kernel<<<BATCH, 256, 0, stream>>>(x, ctx, Wv, bv, Wo, bo, g, be, out);
}
